// TransformerAttention_58712202936594
// MI455X (gfx1250) — matrix operand order repaired and checked by reading, not compiled, not run on hardware
//
#include <hip/hip_runtime.h>

typedef __bf16 bf16;
typedef __attribute__((ext_vector_type(16))) __bf16 v16bf;
typedef __attribute__((ext_vector_type(8)))  __bf16 v8bf;
typedef __attribute__((ext_vector_type(8)))  float  v8f;
typedef int v4i_vec __attribute__((vector_size(16)));
typedef unsigned int u32x4 __attribute__((vector_size(16)));
typedef int i32x8 __attribute__((vector_size(32)));
typedef int i32x4 __attribute__((vector_size(16)));

#define NEG_INF_F (-1000000000.0f)

constexpr int Bb = 4, Ss = 2048, Dd = 1024, Hh = 16, DH = 64;
constexpr int Mrows = Bb * Ss;   // 8192

// ------------------------------------------------------------ async staging
// gfx1250 async global->LDS (ASYNCcnt) with sync fallback.

#if __has_builtin(__builtin_amdgcn_global_load_async_to_lds_b128)
#define HAVE_ASYNC 1
#else
#define HAVE_ASYNC 0
#endif

__device__ __forceinline__ void stage16(void* lds, const void* g) {
#if HAVE_ASYNC
  __builtin_amdgcn_global_load_async_to_lds_b128(
      (__attribute__((address_space(1))) v4i_vec*)g,
      (__attribute__((address_space(3))) v4i_vec*)lds,
      /*offset=*/0, /*cpol=*/0);
#else
  *(v8bf*)lds = *(const v8bf*)g;
#endif
}

__device__ __forceinline__ void wait_stage() {
#if HAVE_ASYNC && __has_builtin(__builtin_amdgcn_s_wait_asynccnt)
  __builtin_amdgcn_s_wait_asynccnt(0);
#endif
}

// ------------------------------------------------------------ TDM staging
// Tensor Data Mover: one descriptor-driven DMA per tile, tracked by TENSORcnt.
// D# packing per CDNA5 ISA.
// This toolchain exposes the 6-arg builtin: (g0, g1, g2, g3, g_extra, cpol).

#if __has_builtin(__builtin_amdgcn_tensor_load_to_lds) && \
    __has_builtin(__builtin_amdgcn_s_wait_tensorcnt)
#define HAVE_TDM 1
#else
#define HAVE_TDM 0
#endif

#if HAVE_TDM
__device__ __forceinline__ unsigned lds_offset(const void* p) {
  return (unsigned)(unsigned long long)
      (__attribute__((address_space(3))) const void*)p;
}

// 2D tile load, bf16 elements. dim0 = contiguous dimension (elements).
// If pad=true: insert 16B of LDS padding after every 64B row (32 elem rows
// -> 80B LDS pitch, matching LDV=40).
__device__ __forceinline__ void tdm_load_2d(unsigned lds_addr, const void* gaddr,
                                            unsigned tensor_d0, unsigned tensor_d1,
                                            unsigned tile_d0, unsigned tile_d1,
                                            unsigned stride0, bool pad) {
  const unsigned long long ga = (unsigned long long)gaddr;
  u32x4 g0;
  g0[0] = 1u;                                   // count=1 (valid user D#)
  g0[1] = lds_addr;                             // lds_addr [63:32]
  g0[2] = (unsigned)(ga & 0xffffffffu);         // global_addr lo
  g0[3] = (unsigned)((ga >> 32) & 0x01ffffffu)  // global_addr [56:32]
          | (2u << 30);                         // type=2 ("image")
  i32x8 g1;
  unsigned w0 = (1u << 16);                     // data_size=1 -> 2 bytes
  if (pad) w0 |= (1u << 20)                     // pad_enable
             |  (3u << 22)                      // pad_interval: 16 DW (64B)
             |  (3u << 25);                     // pad_amount:   4 DW (16B)
  g1[0] = (int)w0;
  g1[1] = (int)((tensor_d0 & 0xffffu) << 16);                    // dim0 lo16
  g1[2] = (int)((tensor_d0 >> 16) | ((tensor_d1 & 0xffffu) << 16));
  g1[3] = (int)((tensor_d1 >> 16) | (tile_d0 << 16));            // tile_dim0
  g1[4] = (int)(tile_d1);                                        // tile_dim1
  g1[5] = (int)(stride0);                                        // dim0 stride
  g1[6] = 0;
  g1[7] = 0;
  i32x4 z4 = {0, 0, 0, 0};
  i32x8 z8 = {0, 0, 0, 0, 0, 0, 0, 0};
  __builtin_amdgcn_tensor_load_to_lds(g0, g1, z4, z4, z8, /*cpol=*/0);
}
#endif

// ------------------------------------------------------------ WMMA fragments
// Contiguous k order (k = 16*half + e) used for A and B alike, not the ISA table order.

// A (16x32 bf16, MxK): lane row m = lane%16, elem e -> k = k0+half*16+e
__device__ __forceinline__ v16bf load_a_frag(const bf16* __restrict__ row,
                                             int half, int k0) {
  v8bf lo = *(const v8bf*)(row + k0 + half * 16);
  v8bf hi = *(const v8bf*)(row + k0 + half * 16 + 8);
  return __builtin_shufflevector(lo, hi, 0, 1, 2, 3, 4, 5, 6, 7,
                                 8, 9, 10, 11, 12, 13, 14, 15);
}

// B (32x16 bf16, KxN): lane col n = lane%16, elem e -> k = half*16+e.
__device__ __forceinline__ v16bf load_b_frag(const bf16* __restrict__ p) {
  v8bf lo = *(const v8bf*)(p);
  v8bf hi = *(const v8bf*)(p + 8);
  return __builtin_shufflevector(lo, hi, 0, 1, 2, 3, 4, 5, 6, 7,
                                 8, 9, 10, 11, 12, 13, 14, 15);
}

__device__ __forceinline__ v8f wmma_bf16(v16bf a, v16bf b, v8f c) {
  return __builtin_amdgcn_wmma_f32_16x16x32_bf16(
      false, a, false, b, (short)0, c, false, false);
}

__device__ __forceinline__ unsigned short to_bf16_raw(float f) {
  return __builtin_bit_cast(unsigned short, (bf16)f);
}

// ------------------------------------------------------------ conversions

__global__ void cvt_f32_to_bf16(const float* __restrict__ src,
                                unsigned short* __restrict__ dst, int n) {
  int i = (blockIdx.x * blockDim.x + threadIdx.x) * 4;
  if (i + 3 < n) {
    float4 v = *(const float4*)(src + i);
    dst[i + 0] = to_bf16_raw(v.x);
    dst[i + 1] = to_bf16_raw(v.y);
    dst[i + 2] = to_bf16_raw(v.z);
    dst[i + 3] = to_bf16_raw(v.w);
  } else {
    for (; i < n; ++i) dst[i] = to_bf16_raw(src[i]);
  }
}

// out[c*R + r] = bf16(in[r*C + c]); 32x32 LDS tile transpose.
__global__ __launch_bounds__(256) void cvt_transpose_bf16(
    const float* __restrict__ in, unsigned short* __restrict__ out, int R, int C) {
  __shared__ float tile[32][33];
  const int tx = threadIdx.x & 31, ty = threadIdx.x >> 5;  // ty 0..7
  const int r0 = blockIdx.x * 32, c0 = blockIdx.y * 32;
#pragma unroll
  for (int i = 0; i < 4; ++i)
    tile[ty + i * 8][tx] = in[(size_t)(r0 + ty + i * 8) * C + c0 + tx];
  __syncthreads();
#pragma unroll
  for (int i = 0; i < 4; ++i)
    out[(size_t)(c0 + ty + i * 8) * R + r0 + tx] =
        to_bf16_raw(tile[tx][ty + i * 8]);
}

// ------------------------------------------------------------ GEMM (WMMA)
// C[M,N] = A[M,K] * W[K,N] + bias, with W supplied TRANSPOSED (WT[N][K]).
// Block: 8 waves, tile 128(M) x 64(N); double-buffered async LDS B tile.
// MODE 0: bf16 out, [B,H,S,64] head layout (Q,K projections)
// MODE 1: fp32 out, row-major                (final projection)
// MODE 2: bf16 out, [B,H,64,S] transposed head layout (V projection)

template <int MODE>
__global__ __launch_bounds__(256) void gemm_wmma(
    const unsigned short* __restrict__ Ap, const unsigned short* __restrict__ WTp,
    const float* __restrict__ bias, unsigned short* __restrict__ outb,
    float* __restrict__ outf) {
  constexpr int K = Dd, N = Dd;
  constexpr int LDB = 40;                       // padded row (80 B)
  const bf16* A  = (const bf16*)Ap;
  const bf16* WT = (const bf16*)WTp;

  __shared__ __align__(128) bf16 BtS[2][64 * LDB];   // [n][k] bf16

  const int tid  = threadIdx.x;
  const int wave = tid >> 5, lane = tid & 31;
  const int half = lane >> 4, ln = lane & 15;
  const int bm = blockIdx.x * 128;
  const int bn = blockIdx.y * 64;

  const bf16* Arow = A + (size_t)(bm + wave * 16 + ln) * K;
  const int nrow = tid >> 2;            // 0..63
  const int kc   = (tid & 3) * 8;       // 0,8,16,24
  const bf16* Wrow = WT + (size_t)(bn + nrow) * K;

  stage16(&BtS[0][nrow * LDB + kc], Wrow + kc);

  v8f c[4] = {{}, {}, {}, {}};
  int buf = 0;

  for (int k0 = 0; k0 < K; k0 += 32) {
    wait_stage();
    __syncthreads();
    if (k0 + 32 < K)
      stage16(&BtS[buf ^ 1][nrow * LDB + kc], Wrow + k0 + 32 + kc);

    const v16bf a = load_a_frag(Arow, half, k0);
#pragma unroll
    for (int j = 0; j < 4; ++j) {
      v16bf b = load_b_frag(&BtS[buf][(j * 16 + ln) * LDB + half * 16]);
      c[j] = wmma_bf16(a, b, c[j]);
    }
    __syncthreads();
    buf ^= 1;
  }

  const int mbase = bm + wave * 16 + half * 8;
#pragma unroll
  for (int j = 0; j < 4; ++j) {
    const int gn = bn + j * 16 + ln;
    const float bv = bias[gn];
#pragma unroll
    for (int r = 0; r < 8; ++r) {
      const int gm = mbase + r;
      const float val = c[j][r] + bv;
      if constexpr (MODE == 1) {
        outf[(size_t)gm * N + gn] = val;
      } else {
        const int b_ = gm / Ss, s_ = gm % Ss;
        const int h_ = gn / DH, f_ = gn % DH;
        size_t idx;
        if constexpr (MODE == 0)
          idx = (((size_t)b_ * Hh + h_) * Ss + s_) * DH + f_;   // [B,H,S,64]
        else
          idx = (((size_t)b_ * Hh + h_) * DH + f_) * Ss + s_;   // [B,H,64,S]
        outb[idx] = to_bf16_raw(val);
      }
    }
  }
}

// ------------------------------------------------------------ flash attention
// grid (S/128, H, B), 256 thr. Wave w owns q rows qblk*128+w*16..+15.
// K streamed as [32 key][64 f] tiles, V as transposed [64 f][32 key] tiles,
// double-buffered. Staging via TDM (one 2D descriptor per tile, wave 0 issues,
// TENSORcnt) when available; per-thread async LDS loads otherwise.

constexpr int LDV = 40;

__device__ __forceinline__ void stage_tiles(bf16* ktile, bf16* vtile,
                                            const bf16* Kb, const bf16* Vb,
                                            int kb, int tid) {
#if HAVE_TDM
  if (tid < 32) {  // wave 0 only (TDM is per-wave, EXEC-independent)
    // K tile: tensor [Ss][DH], tile [32][DH], row-major, no padding
    tdm_load_2d(lds_offset(ktile), Kb + (size_t)kb * DH,
                /*tensor_d0=*/DH, /*tensor_d1=*/Ss,
                /*tile_d0=*/DH, /*tile_d1=*/32, /*stride0=*/DH, false);
    // V^T tile: tensor [DH][Ss], tile [64 f][32 key], pad rows 64B->80B (LDV=40)
    tdm_load_2d(lds_offset(vtile), Vb + kb,
                /*tensor_d0=*/Ss, /*tensor_d1=*/DH,
                /*tile_d0=*/32, /*tile_d1=*/DH, /*stride0=*/Ss, true);
  }
#else
  const int kr = tid >> 3, ch = (tid & 7) * 8;    // K: 32 rows x 8 chunks
  stage16(ktile + kr * 64 + ch, Kb + (size_t)(kb + kr) * DH + ch);
  const int fr = tid >> 2, kc2 = (tid & 3) * 8;   // V: 64 rows x 4 chunks
  stage16(vtile + fr * LDV + kc2, Vb + (size_t)fr * Ss + kb + kc2);
#endif
}

__device__ __forceinline__ void wait_tiles() {
#if HAVE_TDM
  __builtin_amdgcn_s_wait_tensorcnt(0);   // no-op for waves with TENSORcnt==0
#else
  wait_stage();
#endif
}

__global__ __launch_bounds__(256) void flash_attn(
    const unsigned short* __restrict__ Qp, const unsigned short* __restrict__ Kp,
    const unsigned short* __restrict__ VTp, const unsigned char* __restrict__ mask,
    unsigned short* __restrict__ ctxp) {
  const bf16* Q  = (const bf16*)Qp;
  const bf16* K  = (const bf16*)Kp;
  const bf16* VT = (const bf16*)VTp;

  __shared__ __align__(128) bf16 KtS[2][32 * 64];
  __shared__ __align__(128) bf16 VtS[2][64 * LDV];
  __shared__ __align__(128) bf16 PlS[8 * 16 * 32];   // per-wave P tile

  const int tid  = threadIdx.x;
  const int wave = tid >> 5, lane = tid & 31;
  const int half = lane >> 4, ln = lane & 15;
  const int b = blockIdx.z, h = blockIdx.y;
  const int qbase = blockIdx.x * 128 + wave * 16;

  const bf16* Qb = Q  + ((size_t)(b * Hh + h)) * Ss * DH;   // [S][64]
  const bf16* Kb = K  + ((size_t)(b * Hh + h)) * Ss * DH;   // [S][64]
  const bf16* Vb = VT + ((size_t)(b * Hh + h)) * DH * Ss;   // [64][S]
  const unsigned char* km = mask + (size_t)b * Ss;

  // q fragments live in registers for the whole key sweep
  const bf16* qrow = Qb + (size_t)(qbase + ln) * DH;
  const v16bf aq0 = load_a_frag(qrow, half, 0);
  const v16bf aq1 = load_a_frag(qrow, half, 32);

  float mi[8], li[8];
  v8f acc[4] = {{}, {}, {}, {}};
#pragma unroll
  for (int r = 0; r < 8; ++r) { mi[r] = -3.0e38f; li[r] = 0.0f; }

  stage_tiles(&KtS[0][0], &VtS[0][0], Kb, Vb, 0, tid);

  bf16* Pl = &PlS[wave * 16 * 32];
  int buf = 0;

  for (int kb = 0; kb < Ss; kb += 32) {
    wait_tiles();
    __syncthreads();
    if (kb + 32 < Ss)
      stage_tiles(&KtS[buf ^ 1][0], &VtS[buf ^ 1][0], Kb, Vb, kb + 32, tid);

    const bf16* Kt = &KtS[buf][0];
    const bf16* Vt = &VtS[buf][0];

    // ---- scores S[16 q x 32 key] = Q(16x64) * K^T : 4 WMMAs
    v8f s0 = {}, s1 = {};
    {
      v16bf bk = load_b_frag(Kt + ln * 64 + half * 16);        // keys 0..15
      s0 = wmma_bf16(aq0, bk, s0);
      bk = load_b_frag(Kt + ln * 64 + 32 + half * 16);
      s0 = wmma_bf16(aq1, bk, s0);
      bk = load_b_frag(Kt + (16 + ln) * 64 + half * 16);       // keys 16..31
      s1 = wmma_bf16(aq0, bk, s1);
      bk = load_b_frag(Kt + (16 + ln) * 64 + 32 + half * 16);
      s1 = wmma_bf16(aq1, bk, s1);
    }

    const bool km0 = km[kb + ln] != 0;
    const bool km1 = km[kb + 16 + ln] != 0;

    // ---- online softmax; row m = half*8+r lives in the 16 lanes of a half
#pragma unroll
    for (int r = 0; r < 8; ++r) {
      float v0 = km0 ? s0[r] : NEG_INF_F;
      float v1 = km1 ? s1[r] : NEG_INF_F;
      float mx = fmaxf(v0, v1);
#pragma unroll
      for (int off = 1; off < 16; off <<= 1)
        mx = fmaxf(mx, __shfl_xor(mx, off, 32));
      const float mnew = fmaxf(mi[r], mx);
      const float p0 = __expf(v0 - mnew);
      const float p1 = __expf(v1 - mnew);
      float ls = p0 + p1;
#pragma unroll
      for (int off = 1; off < 16; off <<= 1)
        ls += __shfl_xor(ls, off, 32);
      const float alpha = __expf(mi[r] - mnew);
      li[r] = li[r] * alpha + ls;
      mi[r] = mnew;
      const int mrow = half * 8 + r;
      Pl[mrow * 32 + ln]      = (bf16)p0;
      Pl[mrow * 32 + 16 + ln] = (bf16)p1;
#pragma unroll
      for (int j = 0; j < 4; ++j) acc[j][r] = acc[j][r] * alpha;
    }

    // ---- ctx += P(16x32) * V(32x64); Vt is [f][key] so B frags are contiguous
    const v16bf ap = load_a_frag(Pl + ln * 32, half, 0);
#pragma unroll
    for (int j = 0; j < 4; ++j) {
      v16bf bv = load_b_frag(Vt + (j * 16 + ln) * LDV + half * 16);
      acc[j] = wmma_bf16(ap, bv, acc[j]);
    }

    __syncthreads();
    buf ^= 1;
  }

  // ---- normalize, q-row mask, write ctx as [B,S,H*64] bf16
#pragma unroll
  for (int r = 0; r < 8; ++r) {
    const int s_ = qbase + half * 8 + r;
    const float qm = km[s_] ? 1.0f : 0.0f;
    const float inv = qm / li[r];
#pragma unroll
    for (int j = 0; j < 4; ++j) {
      const size_t idx = ((size_t)b * Ss + s_) * Dd + h * DH + j * 16 + ln;
      ctxp[idx] = to_bf16_raw(acc[j][r] * inv);
    }
  }
}

// ------------------------------------------------------------ launch

extern "C" void kernel_launch(void* const* d_in, const int* in_sizes, int n_in,
                              void* d_out, int out_size, void* d_ws, size_t ws_size,
                              hipStream_t stream) {
  (void)in_sizes; (void)n_in; (void)out_size; (void)ws_size;
  const float* X  = (const float*)d_in[0];
  const unsigned char* mask = (const unsigned char*)d_in[1];
  const float* Wq = (const float*)d_in[2];
  const float* bq = (const float*)d_in[3];
  const float* Wk = (const float*)d_in[4];
  const float* bk = (const float*)d_in[5];
  const float* Wv = (const float*)d_in[6];
  const float* bv = (const float*)d_in[7];
  const float* Wo = (const float*)d_in[8];
  const float* bo = (const float*)d_in[9];
  float* out = (float*)d_out;

  const size_t nX = (size_t)Mrows * Dd;
  const size_t nW = (size_t)Dd * Dd;

  unsigned short* Xb  = (unsigned short*)d_ws;
  unsigned short* WqT = Xb  + nX;
  unsigned short* WkT = WqT + nW;
  unsigned short* WvT = WkT + nW;
  unsigned short* WoT = WvT + nW;
  unsigned short* Qb  = WoT + nW;
  unsigned short* Kb  = Qb  + nX;
  unsigned short* Vb  = Kb  + nX;   // [B,H,64,S] (transposed head layout)
  unsigned short* Cb  = Vb  + nX;

  cvt_f32_to_bf16<<<(unsigned)((nX + 1023) / 1024), 256, 0, stream>>>(X, Xb, (int)nX);

  dim3 tgrid(Dd / 32, Dd / 32);                 // 32 x 32
  cvt_transpose_bf16<<<tgrid, 256, 0, stream>>>(Wq, WqT, Dd, Dd);
  cvt_transpose_bf16<<<tgrid, 256, 0, stream>>>(Wk, WkT, Dd, Dd);
  cvt_transpose_bf16<<<tgrid, 256, 0, stream>>>(Wv, WvT, Dd, Dd);
  cvt_transpose_bf16<<<tgrid, 256, 0, stream>>>(Wo, WoT, Dd, Dd);

  dim3 ggrid(Mrows / 128, Dd / 64);             // 64 x 16
  gemm_wmma<0><<<ggrid, 256, 0, stream>>>(Xb, WqT, bq, Qb, nullptr);
  gemm_wmma<0><<<ggrid, 256, 0, stream>>>(Xb, WkT, bk, Kb, nullptr);
  gemm_wmma<2><<<ggrid, 256, 0, stream>>>(Xb, WvT, bv, Vb, nullptr);

  dim3 agrid(Ss / 128, Hh, Bb);                 // 16 x 16 x 4
  flash_attn<<<agrid, 256, 0, stream>>>(Qb, Kb, Vb, mask, Cb);

  gemm_wmma<1><<<ggrid, 256, 0, stream>>>(Cb, WoT, bo, nullptr, out);
}
